// MCA_6768868459106
// MI455X (gfx1250) — compile-verified, matrix operand order checked by reading, not run on hardware
//
#include <hip/hip_runtime.h>
#include <hip/hip_bf16.h>
#include <math.h>

// MI455X / gfx1250, wave32. All heavy math goes through v_wmma_f32_16x16x32_bf16;
// the flash kernel's K-tile staging uses the Tensor Data Mover (tensor_load_to_lds
// + s_wait_tensorcnt) with hardware LDS padding to produce the bank-conflict-free
// 136-element row stride.
// Pipeline:
//   o_rgb  = l2norm(rgb  @ rgb_proj[h])          (GEMM kernel, MODE 0)
//   o_flow = l2norm(flow @ flow_proj[h])         (GEMM kernel, MODE 0)
//   T      = o_rgb @ atte_w[h]                   (GEMM kernel, MODE 1)
//   f_rgb  = tanh(gelu(Flash(Q=T,      K=o_flow, V=o_rgb )) + rgb )
//   f_flow = tanh(gelu(Flash(Q=o_flow, K=T,      V=o_flow)) + flow)
// Flash-style streaming avoids the 512MB atte tensor entirely.

#define N_  4
#define L_  2048
#define D_  1024
#define H_  8
#define DH_ 128

typedef __bf16 bf16_t;
typedef unsigned int       u32;
typedef unsigned long long u64;
typedef __attribute__((ext_vector_type(16))) __bf16 v16bf;
typedef __attribute__((ext_vector_type(8)))  __bf16 v8bf;
typedef __attribute__((ext_vector_type(4)))  __bf16 v4bf;
typedef __attribute__((ext_vector_type(8)))  float  v8f;
typedef __attribute__((ext_vector_type(4)))  u32    v4u;
typedef __attribute__((ext_vector_type(8)))  int    v8i;
typedef __attribute__((ext_vector_type(4)))  int    v4i;

static __device__ __forceinline__ v8f wmma_bf16(v16bf a, v16bf b, v8f c) {
  // 8 args: (neg_a, A, neg_b, B, c_mod, C, reuse_a, reuse_b)
  return __builtin_amdgcn_wmma_f32_16x16x32_bf16(false, a, false, b, (short)0, c,
                                                 false, false);
}

// 16x32 bf16 fragment per ISA 7.12.2:
//   lane<16 : row M=lane,    K = {0..7, 16..23}
//   lane>=16: row M=lane-16, K = {8..15, 24..31}
// base points at a [16 x 32] tile, row-major with `stride` elements per row.
// (B fragments use the same pattern from a [N][K]-transposed LDS tile.)
static __device__ __forceinline__ v16bf load_frag(const bf16_t* base, int stride,
                                                  int lane) {
  const int r  = lane & 15;
  const int kb = (lane & 16) ? 8 : 0;
  const bf16_t* p = base + (size_t)r * stride + kb;
  v8bf lo = *(const v8bf*)(p);
  v8bf hi = *(const v8bf*)(p + 16);
  v16bf f;
#pragma unroll
  for (int i = 0; i < 8; ++i) { f[i] = lo[i]; f[i + 8] = hi[i]; }
  return f;
}

static __device__ __forceinline__ v8f zero8() {
  v8f z = {0.f, 0.f, 0.f, 0.f, 0.f, 0.f, 0.f, 0.f};
  return z;
}

// ---------------------------------------------------------------------------
// GEMM: O[l,k] = A[l,:] @ B[:,k]  per (n,h) slice, 128x128 output per block.
// MODE 0: A = f32 [L_,D_] (shared across heads), B = f32 proj[h] [D_,128],
//         epilogue = per-row L2 normalize. Kdim = 1024.
// MODE 1: A = bf16 o_rgb slice [L_,128], B = f32 atte_w[h] [128,128]. Kdim=128.
// Output bf16 [n,h,l,128].
// ---------------------------------------------------------------------------
template <int MODE>
__global__ __launch_bounds__(256) void mca_gemm_kernel(
    const float* __restrict__ Af, const bf16_t* __restrict__ Ab,
    const float* __restrict__ Bf, bf16_t* __restrict__ O, int Kdim) {
  __shared__ bf16_t As[128 * 40];   // padded stride 40 (80B rows, 16B aligned)
  __shared__ bf16_t Bs[128 * 40];   // stored transposed: Bs[n][k]

  const int nh = blockIdx.y, n = nh / H_, h = nh % H_;
  const int l0 = blockIdx.x * 128;
  const int tid = threadIdx.x, wave = tid >> 5, lane = tid & 31;
  const int half = (lane >> 4) & 1, ln = lane & 15;

  const float*  Ag  = (MODE == 0) ? Af + ((size_t)n * L_ + l0) * (size_t)D_ : nullptr;
  const bf16_t* Ag2 = (MODE == 0) ? nullptr : Ab + ((size_t)nh * L_ + l0) * DH_;
  const float* Bg = Bf + (size_t)h * Kdim * DH_;
  bf16_t* Og = O + ((size_t)nh * L_ + l0) * DH_;

  v8f acc[8];
#pragma unroll
  for (int t = 0; t < 8; ++t) acc[t] = zero8();

  for (int k0 = 0; k0 < Kdim; k0 += 32) {
    __syncthreads();
    // --- Stage A tile: 128 rows x 32 k -----------------------------------
    if (MODE == 0) {
      // f32 -> bf16, float4 loads + packed 8B LDS stores (4096 elems).
#pragma unroll
      for (int j = 0; j < 4; ++j) {
        int e4 = (tid + 256 * j) * 4;          // 0..4092, step 4
        int row = e4 >> 5, k = e4 & 31;
        float4 v = *(const float4*)(Ag + (size_t)row * D_ + k0 + k);
        v4bf d;
        d[0] = (bf16_t)v.x; d[1] = (bf16_t)v.y;
        d[2] = (bf16_t)v.z; d[3] = (bf16_t)v.w;
        *(v4bf*)(&As[row * 40 + k]) = d;
      }
    } else {
      // bf16 copy, b128 loads/stores (4096 elems).
#pragma unroll
      for (int j = 0; j < 2; ++j) {
        int e8 = (tid + 256 * j) * 8;          // 0..4088, step 8
        int row = e8 >> 5, k = e8 & 31;
        *(v8bf*)(&As[row * 40 + k]) =
            *(const v8bf*)(Ag2 + (size_t)row * DH_ + k0 + k);
      }
    }
    // --- Stage B tile transposed: Bs[n][k] = B[k0+k][n] ------------------
#pragma unroll
    for (int j = 0; j < 4; ++j) {
      int e4 = (tid + 256 * j) * 4;            // 0..4092, step 4
      int k = e4 >> 7, col = e4 & 127;
      float4 v = *(const float4*)(Bg + (size_t)(k0 + k) * DH_ + col);
      Bs[(col + 0) * 40 + k] = (bf16_t)v.x;
      Bs[(col + 1) * 40 + k] = (bf16_t)v.y;
      Bs[(col + 2) * 40 + k] = (bf16_t)v.z;
      Bs[(col + 3) * 40 + k] = (bf16_t)v.w;
    }
    __syncthreads();

    v16bf afrag = load_frag(&As[(wave * 16) * 40], 40, lane);
#pragma unroll
    for (int t = 0; t < 8; ++t) {
      v16bf bfrag = load_frag(&Bs[(t * 16) * 40], 40, lane);
      acc[t] = wmma_bf16(afrag, bfrag, acc[t]);
    }
  }

  // Epilogue. C/D layout: VGPR r -> row m = r + 8*half; col = ln + 16*t.
  float inv[8];
#pragma unroll
  for (int r = 0; r < 8; ++r) {
    if (MODE == 0) {
      float s = 0.f;
#pragma unroll
      for (int t = 0; t < 8; ++t) s += acc[t][r] * acc[t][r];
      s += __shfl_xor(s, 1);  s += __shfl_xor(s, 2);
      s += __shfl_xor(s, 4);  s += __shfl_xor(s, 8);   // full-row sum within half
      inv[r] = 1.0f / fmaxf(sqrtf(s), 1e-12f);
    } else {
      inv[r] = 1.0f;
    }
  }
#pragma unroll
  for (int t = 0; t < 8; ++t)
#pragma unroll
    for (int r = 0; r < 8; ++r) {
      int m = wave * 16 + r + 8 * half;
      Og[(size_t)m * DH_ + t * 16 + ln] = (bf16_t)(acc[t][r] * inv[r]);
    }
}

// ---------------------------------------------------------------------------
// Flash attention (no scale) + fused epilogue:
//   out[n, l, h*128+k] = tanh( gelu( softmax_s(Q[l]·K[s]) @ V ) + res[...] )
// Block = 128 query rows x full DH, streams S in 64-wide tiles.
// K tile staged by the Tensor Data Mover (wave 0 issues tensor_load_to_lds,
// hardware pads 16B per 256B row -> 136-element LDS stride); V^T staged
// manually (TDM has no transpose).
// ---------------------------------------------------------------------------
__global__ __launch_bounds__(256) void mca_flash_kernel(
    const bf16_t* __restrict__ Q, const bf16_t* __restrict__ Kx,
    const bf16_t* __restrict__ V, const float* __restrict__ res,
    float* __restrict__ out) {
  __shared__ bf16_t Ks[64 * 136];   // K tile [s][dh], 136 = 128 + 4-DWORD pad
  __shared__ bf16_t Vt[128 * 72];   // V tile transposed [dh][s]
  __shared__ bf16_t Ps[128 * 72];   // P staging (C-layout -> A-layout round trip)

  const int nh = blockIdx.y, n = nh / H_, h = nh % H_;
  const int l0 = blockIdx.x * 128;
  const int tid = threadIdx.x, wave = tid >> 5, lane = tid & 31;
  const int half = (lane >> 4) & 1, ln = lane & 15;

  const bf16_t* Qg = Q  + ((size_t)nh * L_ + l0) * DH_;
  const bf16_t* Kg = Kx + (size_t)nh * L_ * DH_;
  const bf16_t* Vg = V  + (size_t)nh * L_ * DH_;

  // Q fragments for this wave's 16-row strip, held in registers for all s.
  v16bf qf[4];
#pragma unroll
  for (int kk = 0; kk < 4; ++kk)
    qf[kk] = load_frag(Qg + (size_t)(wave * 16) * DH_ + kk * 32, DH_, lane);

  v8f o_acc[8];
#pragma unroll
  for (int t = 0; t < 8; ++t) o_acc[t] = zero8();
  float m_i[8], l_i[8];
#pragma unroll
  for (int r = 0; r < 8; ++r) { m_i[r] = -3.0e38f; l_i[r] = 0.f; }

  for (int s0 = 0; s0 < L_; s0 += 64) {
#if defined(__AMDGCN__) && __has_builtin(__builtin_amdgcn_tensor_load_to_lds)
    // ---- K tile via TDM: 64 rows x 128 bf16, global -> LDS with padding ----
    if (wave == 0) {
      u64 gaddr = (u64)(uintptr_t)(Kg + (size_t)s0 * DH_);  // tile start (bytes)
      u32 lds   = (u32)(uintptr_t)(&Ks[0]);                 // LDS byte address
      v4u g0;
      g0[0] = 1u;                                    // count=1 (user descriptor)
      g0[1] = lds;                                   // lds_addr
      g0[2] = (u32)gaddr;                            // global_addr[31:0]
      g0[3] = (u32)((gaddr >> 32) & 0x1ffffffu) | (2u << 30);  // addr[56:32], type=2
      v8i g1;
      g1[0] = (int)((1u << 16) |                     // data_size = 1 -> 2 bytes
                    (1u << 20) |                     // pad_enable
                    (5u << 22) |                     // pad_interval: 64 DW = 256B
                    (3u << 25));                     // pad_amount: 4 DW = 16B
      g1[1] = (int)(128u << 16);                     // tensor_dim0 = 128 (lo16)
      g1[2] = (int)((u32)L_ << 16);                  // tensor_dim1 = 2048 (lo16)
      g1[3] = (int)(128u << 16);                     // tile_dim0 = 128
      g1[4] = 64;                                    // tile_dim1 = 64, tile_dim2 = 0
      g1[5] = 128;                                   // tensor_dim0_stride = 128
      g1[6] = 0; g1[7] = 0;
      v4i gz4 = {0, 0, 0, 0};                        // 2D: groups 2/3 unused
      v8i gz8 = {0, 0, 0, 0, 0, 0, 0, 0};            // extra group (clang-23 form)
      __builtin_amdgcn_tensor_load_to_lds(g0, g1, gz4, gz4, gz8, 0);
    }
    // ---- V^T tile staged by all waves, overlapped with the K DMA ----
#pragma unroll
    for (int j = 0; j < 4; ++j) {
      int e8 = (tid + 256 * j) * 8;   // 0..8184, step 8
      int s = e8 >> 7, k = e8 & 127;
      v8bf vv = *(const v8bf*)(Vg + (size_t)(s0 + s) * DH_ + k);
#pragma unroll
      for (int i = 0; i < 8; ++i) Vt[(k + i) * 72 + s] = vv[i];
    }
    if (wave == 0) __builtin_amdgcn_s_wait_tensorcnt(0);
#else
    // Fallback: manual K copy + V^T scatter.
#pragma unroll
    for (int j = 0; j < 4; ++j) {
      int e8 = (tid + 256 * j) * 8;
      int s = e8 >> 7, k = e8 & 127;
      v8bf kv = *(const v8bf*)(Kg + (size_t)(s0 + s) * DH_ + k);
      v8bf vv = *(const v8bf*)(Vg + (size_t)(s0 + s) * DH_ + k);
      *(v8bf*)(&Ks[s * 136 + k]) = kv;
#pragma unroll
      for (int i = 0; i < 8; ++i) Vt[(k + i) * 72 + s] = vv[i];
    }
#endif
    if (s0 + 64 < L_) {  // gfx1250 global_prefetch_b8 for the next V tile
      __builtin_prefetch(Vg + (size_t)(s0 + 64) * DH_ + tid * 32, 0, 1);
    }
    __syncthreads();

    // S = Q @ K^T for this wave's 16 rows x 64 s-cols.
    v8f sacc[4];
#pragma unroll
    for (int t = 0; t < 4; ++t) {
      v8f s = zero8();
#pragma unroll
      for (int kk = 0; kk < 4; ++kk)
        s = wmma_bf16(qf[kk], load_frag(&Ks[(t * 16) * 136 + kk * 32], 136, lane), s);
      sacc[t] = s;
    }

    // Online softmax per row r (row m = r + 8*half lives within one 16-lane half).
#pragma unroll
    for (int r = 0; r < 8; ++r) {
      float mx = fmaxf(fmaxf(sacc[0][r], sacc[1][r]), fmaxf(sacc[2][r], sacc[3][r]));
      mx = fmaxf(mx, __shfl_xor(mx, 1)); mx = fmaxf(mx, __shfl_xor(mx, 2));
      mx = fmaxf(mx, __shfl_xor(mx, 4)); mx = fmaxf(mx, __shfl_xor(mx, 8));
      float mnew = fmaxf(m_i[r], mx);
      float corr = __expf(m_i[r] - mnew);
      float rs = 0.f;
#pragma unroll
      for (int t = 0; t < 4; ++t) {
        float p = __expf(sacc[t][r] - mnew);
        sacc[t][r] = p; rs += p;
      }
      rs += __shfl_xor(rs, 1); rs += __shfl_xor(rs, 2);
      rs += __shfl_xor(rs, 4); rs += __shfl_xor(rs, 8);
      l_i[r] = l_i[r] * corr + rs;
      m_i[r] = mnew;
#pragma unroll
      for (int t = 0; t < 8; ++t) o_acc[t][r] *= corr;
    }

    // Stage P (bf16) to this wave's private LDS strip; LDS is in-order per wave.
#pragma unroll
    for (int t = 0; t < 4; ++t)
#pragma unroll
      for (int r = 0; r < 8; ++r)
        Ps[(wave * 16 + r + 8 * half) * 72 + t * 16 + ln] = (bf16_t)sacc[t][r];

    // O += P @ V  (k-dim = 64 s, two 32-wide WMMA steps; B frags from V^T).
#pragma unroll
    for (int kk = 0; kk < 2; ++kk) {
      v16bf pf = load_frag(&Ps[(wave * 16) * 72 + kk * 32], 72, lane);
#pragma unroll
      for (int t = 0; t < 8; ++t) {
        v16bf vf = load_frag(&Vt[(t * 16) * 72 + kk * 32], 72, lane);
        o_acc[t] = wmma_bf16(pf, vf, o_acc[t]);
      }
    }
    __syncthreads();
  }

  // Epilogue: O/l, exact gelu, +residual, tanh, scatter into [n,l,h*128+col].
  float linv[8];
#pragma unroll
  for (int r = 0; r < 8; ++r) linv[r] = 1.0f / l_i[r];
#pragma unroll
  for (int t = 0; t < 8; ++t)
#pragma unroll
    for (int r = 0; r < 8; ++r) {
      int row = l0 + wave * 16 + r + 8 * half;
      int col = t * 16 + ln;
      size_t idx = (size_t)n * L_ * D_ + (size_t)row * D_ + h * DH_ + col;
      float x = o_acc[t][r] * linv[r];
      float g = 0.5f * x * (1.0f + erff(x * 0.70710678118654752f));
      out[idx] = tanhf(g + res[idx]);
    }
}

extern "C" void kernel_launch(void* const* d_in, const int* in_sizes, int n_in,
                              void* d_out, int out_size, void* d_ws, size_t ws_size,
                              hipStream_t stream) {
  const float* rgb       = (const float*)d_in[0];
  const float* flow      = (const float*)d_in[1];
  const float* rgb_proj  = (const float*)d_in[2];
  const float* flow_proj = (const float*)d_in[3];
  const float* atte_w    = (const float*)d_in[4];

  float* f_rgb  = (float*)d_out;
  float* f_flow = f_rgb + (size_t)N_ * L_ * D_;

  const size_t oElems = (size_t)N_ * H_ * L_ * DH_;   // 8,388,608 (16 MB bf16)
  bf16_t* o_rgb  = (bf16_t*)d_ws;
  bf16_t* o_flow = o_rgb + oElems;
  bf16_t* Tq     = o_flow + oElems;                   // o_rgb @ atte_w

  dim3 grid(L_ / 128, N_ * H_);
  dim3 block(256);

  mca_gemm_kernel<0><<<grid, block, 0, stream>>>(rgb,  nullptr, rgb_proj,  o_rgb,  D_);
  mca_gemm_kernel<0><<<grid, block, 0, stream>>>(flow, nullptr, flow_proj, o_flow, D_);
  mca_gemm_kernel<1><<<grid, block, 0, stream>>>(nullptr, o_rgb, atte_w,   Tq,     DH_);

  mca_flash_kernel<<<grid, block, 0, stream>>>(Tq,     o_flow, o_rgb,  rgb,  f_rgb);
  mca_flash_kernel<<<grid, block, 0, stream>>>(o_flow, Tq,     o_flow, flow, f_flow);
}
